// EnergyMACE_36421322670509
// MI455X (gfx1250) — compile-verified, matrix operand order checked by reading, not run on hardware
//
#include <hip/hip_runtime.h>
#include <math.h>

// ---------------- problem constants (from reference) ----------------
#define NN 16384
#define EE 262144
#define SS 10
#define KK 64
#define NB 8
#define MLPH 16
#define FEAT_ELEMS (NN * 9 * KK)   // 9,437,184 floats per buffer

#define LDS_STRIDE 72              // 64 + 8 pad (bf16): 144B rows, conflict-free b128

typedef __attribute__((ext_vector_type(16))) __bf16 v16bf;
typedef __attribute__((ext_vector_type(8)))  __bf16 v8bf;
typedef __attribute__((ext_vector_type(8)))  float  v8f;

__device__ __forceinline__ int l_of(int m) { return (m == 0) ? 0 : ((m < 4) ? 1 : 2); }

// ---------------- init / zero ----------------
__global__ void k_init_feats(const int* __restrict__ species,
                             const float* __restrict__ W_embed,
                             float* __restrict__ feats) {
    int idx = blockIdx.x * blockDim.x + threadIdx.x;
    if (idx >= FEAT_ELEMS) return;
    int n = idx / (9 * KK);
    int rem = idx - n * 9 * KK;
    int m = rem / KK;
    int k = rem - m * KK;
    feats[idx] = (m == 0) ? W_embed[species[n] * KK + k] : 0.0f;
}

__global__ void k_zero(float* __restrict__ buf) {
    int idx = blockIdx.x * blockDim.x + threadIdx.x;
    if (idx < FEAT_ELEMS) buf[idx] = 0.0f;
}

// ---------------- edge message + scatter ----------------
// 64 threads per edge (thread = channel k); 4 edges per 256-thread block.
__global__ void k_edge_scatter(const float* __restrict__ pos,
                               const float* __restrict__ shifts,
                               const int* __restrict__ senders,
                               const int* __restrict__ receivers,
                               const float* __restrict__ feats,   // read m=0 only
                               const float* __restrict__ W_rad_i, // [3][NB][K]
                               float* __restrict__ A) {
    int t = blockIdx.x * blockDim.x + threadIdx.x;
    int e = t >> 6;
    int k = t & 63;
    if (e >= EE) return;

    int sn = senders[e], rn = receivers[e];
    float dx = pos[rn * 3 + 0] - pos[sn * 3 + 0] + shifts[e * 3 + 0];
    float dy = pos[rn * 3 + 1] - pos[sn * 3 + 1] + shifts[e * 3 + 1];
    float dz = pos[rn * 3 + 2] - pos[sn * 3 + 2] + shifts[e * 3 + 2];
    float r = sqrtf(dx * dx + dy * dy + dz * dz);
    float rsafe = (r > 1e-9f) ? r : 1.0f;
    float rinv = 1.0f / rsafe;
    float x = dx * rinv, y = dy * rinv, z = dz * rinv;

    const float s3 = 1.73205081f, s15 = 3.87298335f, s5 = 2.23606798f;
    float Y[9];
    Y[0] = 1.0f;
    Y[1] = s3 * x; Y[2] = s3 * y; Y[3] = s3 * z;
    Y[4] = s15 * x * y; Y[5] = s15 * y * z;
    Y[6] = 0.5f * s5 * (3.0f * z * z - 1.0f);
    Y[7] = s15 * x * z;
    Y[8] = 0.5f * s15 * (x * x - y * y);

    // radial basis with polynomial envelope (p=5)
    float u = r * 0.2f;                       // r / R_MAX
    float u2 = u * u, u4 = u2 * u2, u5 = u4 * u, u6 = u5 * u, u7 = u6 * u;
    float env = (u < 1.0f) ? (1.0f - 21.0f * u5 + 35.0f * u6 - 15.0f * u7) : 0.0f;
    const float c = 0.632455532f;             // sqrt(2/R_MAX)
    const float pi = 3.14159265358979f;
    float rad[NB];
#pragma unroll
    for (int b = 0; b < NB; ++b)
        rad[b] = c * sinf((float)(b + 1) * pi * u) / rsafe * env;

    // Rl[l][k] = sum_b rad[b] * W_rad[l][b][k]
    float Rl[3];
#pragma unroll
    for (int l = 0; l < 3; ++l) {
        float acc = 0.0f;
#pragma unroll
        for (int b = 0; b < NB; ++b)
            acc += rad[b] * W_rad_i[(l * NB + b) * KK + k];
        Rl[l] = acc;
    }

    float hs = feats[(sn * 9 + 0) * KK + k];
    float base = hs * 0.0625f;                // fold 1/AVG_NEIGH
#pragma unroll
    for (int m = 0; m < 9; ++m) {
        float val = base * Y[m] * Rl[l_of(m)];
        atomicAdd(&A[(rn * 9 + m) * KK + k], val);
    }
}

// ---------------- WMMA fragment loaders ----------------
// Both A (16xK rows in LDS, row-major) and B (NxK rows in LDS, i.e. W transposed)
// reduce to the same per-lane pattern: two contiguous 16B chunks -> ds_load_b128.
__device__ __forceinline__ v16bf frag_rowmajor(const __bf16* base, int row, int half, int k0) {
    const v8bf* p0 = (const v8bf*)(base + row * LDS_STRIDE + k0 + half * 8);
    const v8bf* p1 = (const v8bf*)(base + row * LDS_STRIDE + k0 + 16 + half * 8);
    v8bf c0 = *p0;
    v8bf c1 = *p1;
    return __builtin_shufflevector(c0, c1, 0, 1, 2, 3, 4, 5, 6, 7,
                                            8, 9, 10, 11, 12, 13, 14, 15);
}

// ---------------- node update: mix (WMMA) + products + species sc ----------------
// Block: 128 threads = 4 waves; handles 16 nodes; wave jt owns output cols [16jt,16jt+16).
// Reads aggregate A (global) + prev feats; writes new feats INTO the A buffer.
__global__ void __launch_bounds__(128)
k_node_update(float* __restrict__ Abuf,                 // in: aggregate, out: new feats
              const float* __restrict__ featsPrev,
              const int* __restrict__ species,
              const float* __restrict__ W_mix_i,        // [3][K][K]
              const float* __restrict__ W_prod_i,       // [3][3][K]
              const float* __restrict__ W_sc_i) {       // [S][3][K][K]
    __shared__ alignas(16) __bf16 sA[9][16][LDS_STRIDE];        // A tile, row-major (M x K)
    __shared__ alignas(16) __bf16 sWT[3][64][LDS_STRIDE];       // W_mix TRANSPOSED (N x K)

    int n0 = blockIdx.x * 16;
    int tid = threadIdx.x;

    for (int idx = tid; idx < 9 * 16 * 64; idx += 128) {
        int m = idx / 1024, rem = idx - m * 1024;
        int row = rem >> 6, k = rem & 63;
        sA[m][row][k] = (__bf16)Abuf[((n0 + row) * 9 + m) * KK + k];
    }
    for (int idx = tid; idx < 3 * 64 * 64; idx += 128) {
        int l = idx / 4096, rem = idx - l * 4096;
        int k = rem >> 6, j = rem & 63;
        sWT[l][j][k] = (__bf16)W_mix_i[(l * KK + k) * KK + j];   // transpose on store
    }
    __syncthreads();

    int wave = tid >> 5, lane = tid & 31;
    int jt = wave;
    int half = lane >> 4;
    int Mlane = lane & 15;
    int j = jt * 16 + Mlane;

    // A_mixed[m] = A[m] @ W_mix[l(m)]  via bf16 WMMA, f32 accumulate
    v8f Am[9];
#pragma unroll
    for (int m = 0; m < 9; ++m) {
        int l = l_of(m);
        v8f acc = {0.f, 0.f, 0.f, 0.f, 0.f, 0.f, 0.f, 0.f};
#pragma unroll
        for (int kt = 0; kt < 2; ++kt) {
            v16bf a = frag_rowmajor(&sA[m][0][0], Mlane, half, kt * 32);
            v16bf b = frag_rowmajor(&sWT[l][0][0], jt * 16 + Mlane, half, kt * 32);
            acc = __builtin_amdgcn_wmma_f32_16x16x32_bf16(
                false, a, false, b, (short)0, acc, false, false);
        }
        Am[m] = acc;
    }

    // inv[n,j] = sum_m Am^2 ; A0 = Am[0]  (fragment layouts align element-wise)
    v8f inv = {0.f, 0.f, 0.f, 0.f, 0.f, 0.f, 0.f, 0.f};
#pragma unroll
    for (int m = 0; m < 9; ++m) inv += Am[m] * Am[m];
    v8f A0 = Am[0];

#pragma unroll
    for (int m = 0; m < 9; ++m) {
        int l = l_of(m);
        float w1 = W_prod_i[(0 * 3 + l) * KK + j];
        float w2 = W_prod_i[(1 * 3 + l) * KK + j];
        float w3 = W_prod_i[(2 * 3 + l) * KK + j];
        v8f Bm = Am[m] * (w1 + w2 * A0 + w3 * inv);

        // species-dependent self-connection + writeback
#pragma unroll
        for (int r = 0; r < 8; ++r) {
            int M = r + 8 * half;
            int n = n0 + M;
            int s = species[n];
            const float* Wsc = W_sc_i + ((s * 3 + l) * KK) * KK;
            const float* fp = &featsPrev[(n * 9 + m) * KK];
            float dot = 0.0f;
#pragma unroll 8
            for (int k = 0; k < KK; ++k) dot += fp[k] * Wsc[k * KK + j];
            Abuf[(n * 9 + m) * KK + j] = Bm[r] + dot;
        }
    }
}

// ---------------- energy readouts ----------------
__global__ void k_energy0(const float* __restrict__ feats,
                          const float* __restrict__ W_ro0,
                          float* __restrict__ out) {
    int n = blockIdx.x * blockDim.x + threadIdx.x;
    if (n >= NN) return;
    float acc = 0.0f;
    const float* f = &feats[(n * 9 + 0) * KK];
#pragma unroll 8
    for (int k = 0; k < KK; ++k) acc += f[k] * W_ro0[k];
    out[n * 2 + 0] = acc;
}

__global__ void k_energy1(const float* __restrict__ feats,
                          const float* __restrict__ W_m1,  // [K][MLPH]
                          const float* __restrict__ b_m1,  // [MLPH]
                          const float* __restrict__ W_m2,  // [MLPH]
                          float* __restrict__ out) {
    int n = blockIdx.x * blockDim.x + threadIdx.x;
    if (n >= NN) return;
    const float* f = &feats[(n * 9 + 0) * KK];
    float acc = 0.0f;
#pragma unroll
    for (int h = 0; h < MLPH; ++h) {
        float z = b_m1[h];
        for (int k = 0; k < KK; ++k) z += f[k] * W_m1[k * MLPH + h];
        float silu = z / (1.0f + expf(-z));
        acc += silu * W_m2[h];
    }
    out[n * 2 + 1] = acc;
}

// ---------------- host driver ----------------
extern "C" void kernel_launch(void* const* d_in, const int* in_sizes, int n_in,
                              void* d_out, int out_size, void* d_ws, size_t ws_size,
                              hipStream_t stream) {
    const float* positions = (const float*)d_in[0];
    const float* shifts    = (const float*)d_in[1];
    const int*   species   = (const int*)d_in[2];
    const int*   senders   = (const int*)d_in[3];
    const int*   receivers = (const int*)d_in[4];
    const float* W_embed   = (const float*)d_in[5];
    const float* W_rad     = (const float*)d_in[6];
    const float* W_mix     = (const float*)d_in[7];
    const float* W_prod    = (const float*)d_in[8];
    const float* W_sc      = (const float*)d_in[9];
    const float* W_ro0     = (const float*)d_in[10];
    const float* W_m1      = (const float*)d_in[11];
    const float* b_m1      = (const float*)d_in[12];
    const float* W_m2      = (const float*)d_in[13];
    float* out = (float*)d_out;

    float* buf0 = (float*)d_ws;               // feats / A ping-pong
    float* buf1 = buf0 + FEAT_ELEMS;

    const int ZB = (FEAT_ELEMS + 255) / 256;
    const int EB = (EE * 64) / 256;           // 64 threads per edge
    const int NBLK = NN / 16;                 // node-update blocks
    const int NE = (NN + 255) / 256;

    // init feats -> buf0
    k_init_feats<<<ZB, 256, 0, stream>>>(species, W_embed, buf0);

    // ----- layer 0 -----
    k_zero<<<ZB, 256, 0, stream>>>(buf1);
    k_edge_scatter<<<EB, 256, 0, stream>>>(positions, shifts, senders, receivers,
                                           buf0, W_rad + 0 * 3 * NB * KK, buf1);
    k_node_update<<<NBLK, 128, 0, stream>>>(buf1, buf0, species,
                                            W_mix + 0 * 3 * KK * KK,
                                            W_prod + 0 * 9 * KK,
                                            W_sc + 0 * SS * 3 * KK * KK);
    k_energy0<<<NE, 256, 0, stream>>>(buf1, W_ro0, out);

    // ----- layer 1 ----- (feats now live in buf1; buf0 becomes aggregate)
    k_zero<<<ZB, 256, 0, stream>>>(buf0);
    k_edge_scatter<<<EB, 256, 0, stream>>>(positions, shifts, senders, receivers,
                                           buf1, W_rad + 1 * 3 * NB * KK, buf0);
    k_node_update<<<NBLK, 128, 0, stream>>>(buf0, buf1, species,
                                            W_mix + 1 * 3 * KK * KK,
                                            W_prod + 1 * 9 * KK,
                                            W_sc + 1 * SS * 3 * KK * KK);
    k_energy1<<<NE, 256, 0, stream>>>(buf0, W_m1, b_m1, W_m2, out);
}
